// MultiHeadAttention_27127013441932
// MI455X (gfx1250) — matrix operand order repaired and checked by reading, not compiled, not run on hardware
//
#include <hip/hip_runtime.h>

typedef __attribute__((ext_vector_type(16))) _Float16 v16h;
typedef __attribute__((ext_vector_type(8)))  float    v8f;

#define CHN 512
#define NH 8
#define KC 64
#define TT 1024
#define BB 4
#define ROWS (BB * TT)   // 4096
#define BH (BB * NH)     // 32

union V16HU { v16h v; unsigned u[8]; };

// A-operand (16x32 f16) per ISA 7.12.2: lane L -> row M = m0 + (L&15);
// reg r holds K pair: K = kb + ((L>>4)&1)*8 + (r&3)*2 + (r>>2)*16
__device__ inline v16h load_a16(const _Float16* base, int stride, int m0, int kb, int lane) {
  int row  = m0 + (lane & 15);
  int koff = kb + ((lane >> 4) & 1) * 8;
  const _Float16* p = base + (long)row * stride + koff;
  V16HU r;
#pragma unroll
  for (int i = 0; i < 8; ++i) {
    int k = (i & 3) * 2 + (i >> 2) * 16;
    r.u[i] = *(const unsigned*)(p + k);
  }
  return r.v;
}

// B-operand (32x16 f16): lane L -> col N = n0 + (L&15);
// reg r holds K pair: K = kb + ((L>>4)&1)*8 + (r&3)*2 + (r>>2)*16  (read from [N][K] row-major)
__device__ inline v16h load_b16(const _Float16* bt, int stride, int n0, int kb, int lane) {
  int col  = n0 + (lane & 15);
  int koff = kb + ((lane >> 4) & 1) * 8;
  const _Float16* p = bt + (long)col * stride + koff;
  V16HU r;
#pragma unroll
  for (int i = 0; i < 8; ++i) r.u[i] = *(const unsigned*)(p + (i & 3) * 2 + (i >> 2) * 16);
  return r.v;
}

__global__ void k_cvt(const float* __restrict__ s, _Float16* __restrict__ d, int n) {
  int i = blockIdx.x * 256 + threadIdx.x;
  if (i < n) d[i] = (_Float16)s[i];
}

// W [in][out] f32 -> Wt [out][in] f16
__global__ void k_wt(const float* __restrict__ w, _Float16* __restrict__ wt) {
  int i = blockIdx.x * 256 + threadIdx.x;   // 512*512 threads
  int o = i & 511, k = i >> 9;
  wt[(long)o * CHN + k] = (_Float16)w[i];
}

// mode 0: q -> [bh][t][64] scaled by 0.125 ; mode 1: k -> [bh][t][64] ; mode 2: v -> [bh][64][t]
__global__ __launch_bounds__(128) void k_proj(const _Float16* __restrict__ A,
    const _Float16* __restrict__ Bt, const float* __restrict__ bias,
    _Float16* __restrict__ dst, int mode) {
  int lane = threadIdx.x & 31;
  int wave = threadIdx.x >> 5;
  int m0 = blockIdx.x * 16;
  int n0 = (blockIdx.y * 4 + wave) * 16;
  v8f acc = {};
  for (int kb = 0; kb < CHN; kb += 32) {
    v16h a = load_a16(A, CHN, m0, kb, lane);
    v16h b = load_b16(Bt, CHN, n0, kb, lane);
    acc = __builtin_amdgcn_wmma_f32_16x16x32_f16(false, a, false, b, (short)0, acc, false, false);
  }
  int col = n0 + (lane & 15);
  float bv = bias[col];
  int h = col >> 6, d = col & 63;
  int half8 = ((lane >> 4) & 1) * 8;
#pragma unroll
  for (int r = 0; r < 8; ++r) {
    int row = m0 + r + half8;
    int bb = row >> 10, t = row & 1023;
    float v = acc[r] + bv;
    if (mode == 0) v *= 0.125f;
    long idx;
    if (mode == 2) idx = ((long)((bb * NH + h) * KC + d)) * TT + t;
    else           idx = ((long)((bb * NH + h) * TT + t)) * KC + d;
    dst[idx] = (_Float16)v;
  }
}

// band[bh][t][dd] = q_scaled[bh][t][:] . emb_rel_k[dd][:]
__global__ void k_band(const _Float16* __restrict__ q, const float* __restrict__ erk,
                       float* __restrict__ band) {
  int i = blockIdx.x * 256 + threadIdx.x;
  if (i >= BH * TT * 9) return;
  int dd = i % 9;
  int tg = i / 9;
  const _Float16* qp = q + (long)tg * KC;
  const float* e = erk + dd * KC;
  float s = 0.f;
#pragma unroll
  for (int ch = 0; ch < KC; ++ch) s += (float)qp[ch] * e[ch];
  band[i] = s;
}

// flash attention: wave = 16 query rows; loop keys in chunks of 32
__global__ __launch_bounds__(128) void k_attn(const _Float16* __restrict__ q,
    const _Float16* __restrict__ k, const _Float16* __restrict__ vt,
    const float* __restrict__ band, const float* __restrict__ erv,
    _Float16* __restrict__ O) {
  __shared__ _Float16 lds_p[4][16 * 32];
  __shared__ float    lds_o[4][16 * 64];
  __shared__ float    lds_m[4][16];
  __shared__ float    lds_l[4][16];
  __shared__ float    lds_pb[4][16 * 9];

  int lane = threadIdx.x & 31;
  int wave = threadIdx.x >> 5;
  int bh = blockIdx.y;
  int qt0 = (blockIdx.x * 4 + wave) * 16;
  const _Float16* qh = q  + (long)bh * TT * KC;
  const _Float16* kh = k  + (long)bh * TT * KC;
  const _Float16* vh = vt + (long)bh * KC * TT;
  const float* bandh = band + (long)bh * TT * 9;

  v16h aq0 = load_a16(qh, KC, qt0, 0, lane);
  v16h aq1 = load_a16(qh, KC, qt0, 32, lane);

  v8f acc0 = {}, acc1 = {}, acc2 = {}, acc3 = {};
  float m[8], l[8];
#pragma unroll
  for (int r = 0; r < 8; ++r) { m[r] = -1e30f; l[r] = 0.f; }
  int half8 = ((lane >> 4) & 1) * 8;
  int col = lane & 15;

  for (int kt = 0; kt < TT; kt += 32) {
    v8f s0 = {}, s1 = {};
    {
      v16h b0 = load_b16(kh, KC, kt, 0, lane);
      s0 = __builtin_amdgcn_wmma_f32_16x16x32_f16(false, aq0, false, b0, (short)0, s0, false, false);
      v16h b1 = load_b16(kh, KC, kt, 32, lane);
      s0 = __builtin_amdgcn_wmma_f32_16x16x32_f16(false, aq1, false, b1, (short)0, s0, false, false);
      v16h b2 = load_b16(kh, KC, kt + 16, 0, lane);
      s1 = __builtin_amdgcn_wmma_f32_16x16x32_f16(false, aq0, false, b2, (short)0, s1, false, false);
      v16h b3 = load_b16(kh, KC, kt + 16, 32, lane);
      s1 = __builtin_amdgcn_wmma_f32_16x16x32_f16(false, aq1, false, b3, (short)0, s1, false, false);
    }
    // banded relative-position bias (nonzero only near the diagonal)
    if (kt + 31 >= qt0 - 4 && kt <= qt0 + 19) {
#pragma unroll
      for (int r = 0; r < 8; ++r) {
        int t = qt0 + r + half8;
        int dd0 = kt + col - t + 4;
        if (dd0 >= 0 && dd0 < 9) s0[r] += bandh[t * 9 + dd0];
        int dd1 = dd0 + 16;
        if (dd1 >= 0 && dd1 < 9) s1[r] += bandh[t * 9 + dd1];
      }
    }
    // online softmax update; rows live across 16-lane groups in C-layout
#pragma unroll
    for (int r = 0; r < 8; ++r) {
      float mr = fmaxf(s0[r], s1[r]);
      mr = fmaxf(mr, __shfl_xor(mr, 1, 16));
      mr = fmaxf(mr, __shfl_xor(mr, 2, 16));
      mr = fmaxf(mr, __shfl_xor(mr, 4, 16));
      mr = fmaxf(mr, __shfl_xor(mr, 8, 16));
      float mn = fmaxf(m[r], mr);
      float sc = __expf(m[r] - mn);
      m[r] = mn;
      float p0 = __expf(s0[r] - mn);
      float p1 = __expf(s1[r] - mn);
      s0[r] = p0; s1[r] = p1;
      float rs = p0 + p1;
      rs += __shfl_xor(rs, 1, 16);
      rs += __shfl_xor(rs, 2, 16);
      rs += __shfl_xor(rs, 4, 16);
      rs += __shfl_xor(rs, 8, 16);
      l[r] = l[r] * sc + rs;
      acc0[r] *= sc; acc1[r] *= sc; acc2[r] *= sc; acc3[r] *= sc;
    }
    // C-layout -> A-layout transpose of P through per-wave LDS
#pragma unroll
    for (int r = 0; r < 8; ++r) {
      int t = r + half8;
      lds_p[wave][t * 32 + col]      = (_Float16)s0[r];
      lds_p[wave][t * 32 + 16 + col] = (_Float16)s1[r];
    }
    __syncthreads();
    v16h ap = load_a16(&lds_p[wave][0], 32, 0, 0, lane);
    v16h bv0 = load_b16(vh, TT, 0, kt, lane);
    acc0 = __builtin_amdgcn_wmma_f32_16x16x32_f16(false, ap, false, bv0, (short)0, acc0, false, false);
    v16h bv1 = load_b16(vh, TT, 16, kt, lane);
    acc1 = __builtin_amdgcn_wmma_f32_16x16x32_f16(false, ap, false, bv1, (short)0, acc1, false, false);
    v16h bv2 = load_b16(vh, TT, 32, kt, lane);
    acc2 = __builtin_amdgcn_wmma_f32_16x16x32_f16(false, ap, false, bv2, (short)0, acc2, false, false);
    v16h bv3 = load_b16(vh, TT, 48, kt, lane);
    acc3 = __builtin_amdgcn_wmma_f32_16x16x32_f16(false, ap, false, bv3, (short)0, acc3, false, false);
    __syncthreads();
  }

  // normalized output tile -> LDS, plus per-row m/l
#pragma unroll
  for (int r = 0; r < 8; ++r) {
    int t = r + half8;
    float inv = 1.f / l[r];
    lds_o[wave][t * 64 + 0 * 16 + col] = acc0[r] * inv;
    lds_o[wave][t * 64 + 1 * 16 + col] = acc1[r] * inv;
    lds_o[wave][t * 64 + 2 * 16 + col] = acc2[r] * inv;
    lds_o[wave][t * 64 + 3 * 16 + col] = acc3[r] * inv;
    if (col == 0) { lds_m[wave][t] = m[r]; lds_l[wave][t] = l[r]; }
  }
  __syncthreads();

  // band probabilities P[t][t+dd-4] for the rel-v correction (9 per row)
  for (int idx = lane; idx < 16 * 9; idx += 32) {
    int tl = idx / 9, dd = idx % 9;
    int t = qt0 + tl;
    int s = t + dd - 4;
    float pb = 0.f;
    if (s >= 0 && s < TT) {
      const _Float16* qp = qh + (long)t * KC;
      const _Float16* kp = kh + (long)s * KC;
      float dot = 0.f;
#pragma unroll
      for (int ch = 0; ch < KC; ++ch) dot += (float)qp[ch] * (float)kp[ch];
      dot += bandh[t * 9 + dd];
      pb = __expf(dot - lds_m[wave][tl]) / lds_l[wave][tl];
    }
    lds_pb[wave][idx] = pb;
  }
  __syncthreads();

  int bb = bh >> 3, h = bh & 7;
  for (int idx = lane; idx < 16 * 64; idx += 32) {
    int tl = idx >> 6, d = idx & 63;
    float corr = 0.f;
#pragma unroll
    for (int dd = 0; dd < 9; ++dd) corr += lds_pb[wave][tl * 9 + dd] * erv[dd * KC + d];
    float val = lds_o[wave][idx] + corr;
    int t = qt0 + tl;
    O[((long)(bb * TT + t)) * CHN + h * KC + d] = (_Float16)val;
  }
}

__global__ __launch_bounds__(128) void k_out(const _Float16* __restrict__ A,
    const _Float16* __restrict__ Bt, const float* __restrict__ bias,
    float* __restrict__ Y) {
  int lane = threadIdx.x & 31;
  int wave = threadIdx.x >> 5;
  int m0 = blockIdx.x * 16;
  int n0 = (blockIdx.y * 4 + wave) * 16;
  v8f acc = {};
  for (int kb = 0; kb < CHN; kb += 32) {
    v16h a = load_a16(A, CHN, m0, kb, lane);
    v16h b = load_b16(Bt, CHN, n0, kb, lane);
    acc = __builtin_amdgcn_wmma_f32_16x16x32_f16(false, a, false, b, (short)0, acc, false, false);
  }
  int col = n0 + (lane & 15);
  float bv = bias[col];
  int half8 = ((lane >> 4) & 1) * 8;
#pragma unroll
  for (int r = 0; r < 8; ++r) {
    int row = m0 + r + half8;
    Y[(long)row * CHN + col] = acc[r] + bv;
  }
}

extern "C" void kernel_launch(void* const* d_in, const int* in_sizes, int n_in,
                              void* d_out, int out_size, void* d_ws, size_t ws_size,
                              hipStream_t stream) {
  (void)in_sizes; (void)n_in; (void)out_size; (void)ws_size;
  const float* x   = (const float*)d_in[0];
  const float* c   = (const float*)d_in[1];
  const float* Wq  = (const float*)d_in[2];
  const float* bq  = (const float*)d_in[3];
  const float* Wk  = (const float*)d_in[4];
  const float* bk  = (const float*)d_in[5];
  const float* Wv  = (const float*)d_in[6];
  const float* bv  = (const float*)d_in[7];
  const float* Wo  = (const float*)d_in[8];
  const float* bo  = (const float*)d_in[9];
  const float* erk = (const float*)d_in[10];
  const float* erv = (const float*)d_in[11];

  char* ws = (char*)d_ws;
  constexpr size_t SZ_ROW  = (size_t)ROWS * CHN * 2;   // 4 MiB f16 buffers
  constexpr size_t SZ_W    = (size_t)CHN * CHN * 2;
  constexpr size_t OFF_XH  = 0;
  constexpr size_t OFF_CH  = OFF_XH + SZ_ROW;
  constexpr size_t OFF_WQ  = OFF_CH + SZ_ROW;
  constexpr size_t OFF_WK  = OFF_WQ + SZ_W;
  constexpr size_t OFF_WV  = OFF_WK + SZ_W;
  constexpr size_t OFF_WO  = OFF_WV + SZ_W;
  constexpr size_t OFF_Q   = OFF_WO + SZ_W;
  constexpr size_t OFF_K   = OFF_Q + SZ_ROW;
  constexpr size_t OFF_V   = OFF_K + SZ_ROW;
  constexpr size_t OFF_BAND= OFF_V + SZ_ROW;
  constexpr size_t SZ_BAND = (size_t)BH * TT * 9 * 4;
  constexpr size_t OFF_O   = OFF_BAND + SZ_BAND;

  _Float16* xh  = (_Float16*)(ws + OFF_XH);
  _Float16* chh = (_Float16*)(ws + OFF_CH);
  _Float16* wqT = (_Float16*)(ws + OFF_WQ);
  _Float16* wkT = (_Float16*)(ws + OFF_WK);
  _Float16* wvT = (_Float16*)(ws + OFF_WV);
  _Float16* woT = (_Float16*)(ws + OFF_WO);
  _Float16* qf  = (_Float16*)(ws + OFF_Q);
  _Float16* kf  = (_Float16*)(ws + OFF_K);
  _Float16* vtf = (_Float16*)(ws + OFF_V);
  float*    bnd = (float*)(ws + OFF_BAND);
  _Float16* of  = (_Float16*)(ws + OFF_O);

  const int NEL = ROWS * CHN;  // 2,097,152
  k_cvt<<<dim3(NEL / 256), 256, 0, stream>>>(x, xh, NEL);
  k_cvt<<<dim3(NEL / 256), 256, 0, stream>>>(c, chh, NEL);
  k_wt<<<dim3(CHN * CHN / 256), 256, 0, stream>>>(Wq, wqT);
  k_wt<<<dim3(CHN * CHN / 256), 256, 0, stream>>>(Wk, wkT);
  k_wt<<<dim3(CHN * CHN / 256), 256, 0, stream>>>(Wv, wvT);
  k_wt<<<dim3(CHN * CHN / 256), 256, 0, stream>>>(Wo, woT);

  dim3 gemmGrid(ROWS / 16, CHN / 64);  // 4 waves/block * 16 cols
  k_proj<<<gemmGrid, 128, 0, stream>>>(xh,  wqT, bq, qf,  0);
  k_proj<<<gemmGrid, 128, 0, stream>>>(chh, wkT, bk, kf,  1);
  k_proj<<<gemmGrid, 128, 0, stream>>>(chh, wvT, bv, vtf, 2);

  k_band<<<dim3((BH * TT * 9 + 255) / 256), 256, 0, stream>>>(qf, erk, bnd);

  k_attn<<<dim3(TT / 64, BH), 128, 0, stream>>>(qf, kf, vtf, bnd, erv, of);

  k_out<<<gemmGrid, 128, 0, stream>>>(of, woT, bo, (float*)d_out);
}
